// QWen3MoeDecoderLayer_87454124081329
// MI455X (gfx1250) — compile-verified, matrix operand order checked by reading, not run on hardware
//
#include <hip/hip_runtime.h>
#include <hip/hip_bf16.h>

#define T_TOK 1024
#define HID   2048
#define NH    32
#define NKV   4
#define HD    128
#define NEXP  64
#define KTOP  8
#define IDIM  768
#define CAP   256
#define REP   (NH / NKV)
#define EPSV  1e-6f
#define THETA 1000000.0f
#define NEGINF (-1e30f)

typedef __bf16 bf16_t;
typedef __attribute__((ext_vector_type(16))) __bf16 v16bf;
typedef __attribute__((ext_vector_type(8)))  __bf16 v8bf;
typedef __attribute__((ext_vector_type(4)))  __bf16 v4bf;
typedef __attribute__((ext_vector_type(2)))  __bf16 v2bf;
typedef __attribute__((ext_vector_type(8)))  float  v8f;

#define BM 64
#define BN 64
#define BK 32
#define LDST 40   // LDS row stride in bf16 elems (80B: 16B-aligned rows, skewed banks)

__device__ __forceinline__ v8f wmma_bf16(v16bf a, v16bf b, v8f c) {
  return __builtin_amdgcn_wmma_f32_16x16x32_bf16(false, a, false, b, (short)0, c,
                                                 false, false);
}

// Fragment from an LDS row of 32 contiguous bf16 (one K-slice).
// lanes 0-15: K = {0..7,16..23}; lanes 16-31: K = {8..15,24..31}  (kb = 0 / 8)
__device__ __forceinline__ v16bf frag_from_lds(const bf16_t* row, int kb) {
  v8bf lo = *(const v8bf*)(row + kb);
  v8bf hi = *(const v8bf*)(row + kb + 16);
  v16bf f;
#pragma unroll
  for (int i = 0; i < 8; ++i) { f[i] = lo[i]; f[i + 8] = hi[i]; }
  return f;
}

// Same fragment pattern sourced from 32 contiguous f32 in global memory.
__device__ __forceinline__ v16bf frag_from_global_f32(const float* p, int kb) {
  float4 a = *(const float4*)(p + kb);
  float4 b = *(const float4*)(p + kb + 4);
  float4 c = *(const float4*)(p + kb + 16);
  float4 d = *(const float4*)(p + kb + 20);
  v16bf f;
  f[0] = (bf16_t)a.x; f[1] = (bf16_t)a.y; f[2]  = (bf16_t)a.z; f[3]  = (bf16_t)a.w;
  f[4] = (bf16_t)b.x; f[5] = (bf16_t)b.y; f[6]  = (bf16_t)b.z; f[7]  = (bf16_t)b.w;
  f[8] = (bf16_t)c.x; f[9] = (bf16_t)c.y; f[10] = (bf16_t)c.z; f[11] = (bf16_t)c.w;
  f[12]= (bf16_t)d.x; f[13]= (bf16_t)d.y; f[14] = (bf16_t)d.z; f[15] = (bf16_t)d.w;
  return f;
}

__device__ __forceinline__ v4bf pack4(float4 x) {
  v4bf p; p[0] = (bf16_t)x.x; p[1] = (bf16_t)x.y; p[2] = (bf16_t)x.z; p[3] = (bf16_t)x.w;
  return p;
}

// ---------------------------------------------------------------- RMSNorm ---
__global__ void rmsnorm_kernel(const float* __restrict__ x, const float* __restrict__ w,
                               float* __restrict__ y, int cols) {
  int row = blockIdx.x;
  const float* xr = x + (long)row * cols;
  float ss = 0.f;
  for (int c = threadIdx.x; c < cols; c += 256) { float v = xr[c]; ss += v * v; }
#pragma unroll
  for (int m = 16; m > 0; m >>= 1) ss += __shfl_xor(ss, m, 32);
  __shared__ float sred[8];
  int wid = threadIdx.x >> 5, lid = threadIdx.x & 31;
  if (lid == 0) sred[wid] = ss;
  __syncthreads();
  float tot = 0.f;
#pragma unroll
  for (int i = 0; i < 8; ++i) tot += sred[i];
  float inv = rsqrtf(tot / (float)cols + EPSV);
  for (int c = threadIdx.x; c < cols; c += 256)
    y[(long)row * cols + c] = xr[c] * inv * w[c];
}

// --------------------------------------------------- generic bf16-WMMA GEMM -
// C[M,N] = A[M,K]*B[K,N] (+Csrc). 64x64 block tile, 8 waves, each wave 32x16
// (2 WMMAs/K-step). Register double-buffered: tile k+1 global loads issue
// between the barriers while tile k's WMMAs run.
// M%64==0, N%64==0, K%32==0. Optional batching via blockIdx.z strides.
__global__ void gemm_kernel(const float* __restrict__ A, const float* __restrict__ B,
                            const float* __restrict__ Csrc, float* __restrict__ C,
                            int M, int N, int Kd, long sA, long sB, long sC) {
  int bz = blockIdx.z;
  A += (long)bz * sA; B += (long)bz * sB; C += (long)bz * sC;
  int m0 = blockIdx.y * BM;
  int n0 = blockIdx.x * BN;
  __shared__ __align__(16) bf16_t As[BM][LDST];
  __shared__ __align__(16) bf16_t Bs[BN][LDST];
  int tid = threadIdx.x;
  int wid = tid >> 5, lane = tid & 31;
  int wm = (wid >> 2) * 32;     // 0 / 32
  int wn = (wid & 3) * 16;      // 0..48
  int lrow = lane & 15;
  int kb = (lane < 16) ? 0 : 8;

  // Per-thread staging coords (constant LDS targets, strided global pointers).
  int ar = tid >> 3;                 // 0..31
  int ac = (tid & 7) * 4;            // 0..28
  const float* ap0 = A + (long)(m0 + ar) * Kd + ac;
  const float* ap1 = A + (long)(m0 + ar + 32) * Kd + ac;
  int bk = (tid >> 4) * 2;           // 0..30 (even)
  int bn = (tid & 15) * 4;           // 0..60
  const float* bp0 = B + (long)bk * N + n0 + bn;
  const float* bp1 = bp0 + N;
  const long bstep = (long)BK * N;

  v8f acc0 = {0.f,0.f,0.f,0.f,0.f,0.f,0.f,0.f};
  v8f acc1 = {0.f,0.f,0.f,0.f,0.f,0.f,0.f,0.f};

  // pipeline prologue: tile 0 into registers
  float4 a0 = *(const float4*)ap0;
  float4 a1 = *(const float4*)ap1;
  float4 b0 = *(const float4*)bp0;
  float4 b1 = *(const float4*)bp1;
  ap0 += BK; ap1 += BK; bp0 += bstep; bp1 += bstep;

  for (int kk = 0; kk < Kd; kk += BK) {
    // commit current tile to LDS (bf16)
    *(v4bf*)&As[ar][ac]      = pack4(a0);
    *(v4bf*)&As[ar + 32][ac] = pack4(a1);
    float e0[4] = {b0.x, b0.y, b0.z, b0.w};
    float e1[4] = {b1.x, b1.y, b1.z, b1.w};
#pragma unroll
    for (int j = 0; j < 4; ++j) {
      v2bf p; p[0] = (bf16_t)e0[j]; p[1] = (bf16_t)e1[j];
      *(v2bf*)&Bs[bn + j][bk] = p;               // transposed, k-pair packed
    }
    __syncthreads();
    // issue next tile's loads; consumed only at next iteration's commit
    if (kk + BK < Kd) {
      a0 = *(const float4*)ap0;
      a1 = *(const float4*)ap1;
      b0 = *(const float4*)bp0;
      b1 = *(const float4*)bp1;
      ap0 += BK; ap1 += BK; bp0 += bstep; bp1 += bstep;
      __builtin_prefetch(bp0, 0, 1);             // tile k+2
    }
    v16bf af0 = frag_from_lds(&As[wm + lrow][0], kb);
    v16bf af1 = frag_from_lds(&As[wm + 16 + lrow][0], kb);
    v16bf bf  = frag_from_lds(&Bs[wn + lrow][0], kb);
    acc0 = wmma_bf16(af0, bf, acc0);
    acc1 = wmma_bf16(af1, bf, acc1);
    __syncthreads();
  }
  int cn = n0 + wn + lrow;
  int rb0 = m0 + wm + ((lane < 16) ? 0 : 8);
  int rb1 = rb0 + 16;
  if (Csrc) {
#pragma unroll
    for (int r = 0; r < 8; ++r) {
      long i0 = (long)(rb0 + r) * N + cn;
      long i1 = (long)(rb1 + r) * N + cn;
      C[i0] = acc0[r] + Csrc[i0];
      C[i1] = acc1[r] + Csrc[i1];
    }
  } else {
#pragma unroll
    for (int r = 0; r < 8; ++r) {
      C[(long)(rb0 + r) * N + cn] = acc0[r];
      C[(long)(rb1 + r) * N + cn] = acc1[r];
    }
  }
}

// ---------------------------------------------- per-head RMSNorm + RoPE ------
__global__ void qknorm_rope_kernel(float* __restrict__ x, const float* __restrict__ w,
                                   const int* __restrict__ pos, int nheads) {
  int t = blockIdx.x, hh = blockIdx.y;
  float* row = x + ((long)t * nheads + hh) * HD;
  int d = threadIdx.x;       // 0..127
  float v = row[d];
  float ss = v * v;
#pragma unroll
  for (int m = 16; m > 0; m >>= 1) ss += __shfl_xor(ss, m, 32);
  __shared__ float sred[4];
  int wid = d >> 5, lid = d & 31;
  if (lid == 0) sred[wid] = ss;
  __syncthreads();
  float tot = sred[0] + sred[1] + sred[2] + sred[3];
  float nv = v * rsqrtf(tot / (float)HD + EPSV) * w[d];
  __shared__ float sh[HD];
  sh[d] = nv;
  __syncthreads();
  const int half = HD / 2;
  float p = (float)pos[t];
  if (d < half) {
    float fr = p * __powf(THETA, -((float)d) / (float)half);
    float cs = __cosf(fr), sn = __sinf(fr);
    float x1 = sh[d], x2 = sh[d + half];
    row[d]        = x1 * cs - x2 * sn;
    row[d + half] = x1 * sn + x2 * cs;
  }
}

// ------------------------------------------ one-wave flash attention tile ----
// grid (NH, T/16), block 32 (one wave). GQA 8:1.
__global__ void attn_kernel(const float* __restrict__ q, const float* __restrict__ k,
                            const float* __restrict__ v, float* __restrict__ o) {
  int hq = blockIdx.x;
  int q0 = blockIdx.y * 16;
  int hkv = hq / REP;
  int lane = threadIdx.x;
  int lrow = lane & 15;
  int kb = (lane < 16) ? 0 : 8;
  const float scale = 0.08838834764831845f;   // 128^-0.5

  v16bf qf[4];
  {
    const float* qrow = q + ((long)(q0 + lrow) * NH + hq) * HD;
#pragma unroll
    for (int c = 0; c < 4; ++c) qf[c] = frag_from_global_f32(qrow + c * 32, kb);
  }

  v8f ofrag[8];
#pragma unroll
  for (int n = 0; n < 8; ++n) ofrag[n] = (v8f){0.f,0.f,0.f,0.f,0.f,0.f,0.f,0.f};
  float mrow[8], lsum[8];
#pragma unroll
  for (int r = 0; r < 8; ++r) { mrow[r] = NEGINF; lsum[r] = 0.f; }

  __shared__ __align__(16) bf16_t Vs[HD][LDST];   // [hd][key] transposed
  __shared__ __align__(16) bf16_t Ps[16][LDST];

  const float* vbase = v + (long)hkv * HD;
  const float* kbase = k + (long)hkv * HD;

  for (int j = 0; j < q0 + 16; j += 32) {
    // stage 32x128 V tile (bf16, transposed): batch 8 float4 loads per burst
    {
      const float4* v4p = (const float4*)(vbase + (long)(j + lane) * NKV * HD);
#pragma unroll
      for (int ch = 0; ch < 4; ++ch) {
        float4 xs[8];
#pragma unroll
        for (int u = 0; u < 8; ++u) xs[u] = v4p[ch * 8 + u];
#pragma unroll
        for (int u = 0; u < 8; ++u) {
          int d = (ch * 8 + u) * 4;
          Vs[d + 0][lane] = (bf16_t)xs[u].x;
          Vs[d + 1][lane] = (bf16_t)xs[u].y;
          Vs[d + 2][lane] = (bf16_t)xs[u].z;
          Vs[d + 3][lane] = (bf16_t)xs[u].w;
        }
      }
    }
    // S = Q K^T for two 16-key subtiles
    v8f s[2];
#pragma unroll
    for (int cc = 0; cc < 2; ++cc) {
      v8f sacc = (v8f){0.f,0.f,0.f,0.f,0.f,0.f,0.f,0.f};
      const float* krow = kbase + (long)(j + cc * 16 + lrow) * NKV * HD;
#pragma unroll
      for (int c = 0; c < 4; ++c) {
        v16bf bf = frag_from_global_f32(krow + c * 32, kb);  // B lane = key col
        sacc = wmma_bf16(qf[c], bf, sacc);
      }
      s[cc] = sacc;
    }
    // causal mask + online softmax (row reductions across half-wave of 16)
#pragma unroll
    for (int r = 0; r < 8; ++r) {
      int row = q0 + r + ((lane < 16) ? 0 : 8);
      int col0 = j + lrow;
      float s0 = s[0][r] * scale; if (col0 > row)      s0 = NEGINF;
      float s1 = s[1][r] * scale; if (col0 + 16 > row) s1 = NEGINF;
      float mx = fmaxf(s0, s1);
#pragma unroll
      for (int m = 8; m > 0; m >>= 1) mx = fmaxf(mx, __shfl_xor(mx, m, 16));
      float mnew = fmaxf(mrow[r], mx);
      float corr = __expf(mrow[r] - mnew);
      float e0 = __expf(s0 - mnew);
      float e1 = __expf(s1 - mnew);
      float ls = e0 + e1;
#pragma unroll
      for (int m = 8; m > 0; m >>= 1) ls += __shfl_xor(ls, m, 16);
      lsum[r] = lsum[r] * corr + ls;
      mrow[r] = mnew;
#pragma unroll
      for (int n = 0; n < 8; ++n) ofrag[n][r] *= corr;
      int rl = r + ((lane < 16) ? 0 : 8);
      Ps[rl][lrow]      = (bf16_t)e0;
      Ps[rl][16 + lrow] = (bf16_t)e1;
    }
    __syncthreads();
    v16bf pf = frag_from_lds(&Ps[lrow][0], kb);      // A frag: P 16x32
#pragma unroll
    for (int n = 0; n < 8; ++n) {
      v16bf vf = frag_from_lds(&Vs[n * 16 + lrow][0], kb);  // B frag: V 32x16
      ofrag[n] = wmma_bf16(pf, vf, ofrag[n]);
    }
    __syncthreads();
  }
#pragma unroll
  for (int n = 0; n < 8; ++n)
#pragma unroll
    for (int r = 0; r < 8; ++r) {
      int row = q0 + r + ((lane < 16) ? 0 : 8);
      o[((long)row * NH + hq) * HD + n * 16 + lrow] = ofrag[n][r] / lsum[r];
    }
}

// ------------------------------------------------ router: top-8 of 64 -------
__global__ void topk_kernel(const float* __restrict__ logits,
                            float* __restrict__ topw, int* __restrict__ topi) {
  int t = blockIdx.x * blockDim.x + threadIdx.x;
  if (t >= T_TOK) return;
  const float* lr = logits + (long)t * NEXP;
  float bv[KTOP]; int bi[KTOP];
#pragma unroll
  for (int i = 0; i < KTOP; ++i) { bv[i] = NEGINF; bi[i] = -1; }
  for (int e = 0; e < NEXP; ++e) {
    float v = lr[e];
    if (v > bv[KTOP - 1]) {
      int jx = KTOP - 1;
      while (jx > 0 && v > bv[jx - 1]) { bv[jx] = bv[jx - 1]; bi[jx] = bi[jx - 1]; --jx; }
      bv[jx] = v; bi[jx] = e;
    }
  }
  // softmax constant cancels in the top-k renormalization
  float mx = bv[0], psum = 0.f;
  float pv[KTOP];
#pragma unroll
  for (int i = 0; i < KTOP; ++i) { pv[i] = __expf(bv[i] - mx); psum += pv[i]; }
#pragma unroll
  for (int i = 0; i < KTOP; ++i) {
    topw[t * KTOP + i] = pv[i] / psum;
    topi[t * KTOP + i] = bi[i];
  }
}

__global__ void init_i32_kernel(int* __restrict__ p, int n, int val) {
  int i = blockIdx.x * 256 + threadIdx.x;
  if (i < n) p[i] = val;
}

// deterministic rank = #earlier flat assignments with same expert (stable sort)
__global__ void rank_kernel(const int* __restrict__ topi, int* __restrict__ slotv,
                            int* __restrict__ tok4slot) {
  int i = blockIdx.x * 256 + threadIdx.x;
  if (i >= T_TOK * KTOP) return;
  int e = topi[i];
  int r = 0;
  for (int jx = 0; jx < i; ++jx) r += (topi[jx] == e);
  if (r < CAP) {
    int s = e * CAP + r;
    slotv[i] = s;
    tok4slot[s] = i / KTOP;
  } else slotv[i] = -1;
}

// --------------------------- fused gate/up expert GEMM with SiLU epilogue ----
// grid (IDIM/64, CAP/64, NEXP). Each wave: 32x16 per matrix -> 4 WMMAs/K-step.
// Register double-buffered like gemm_kernel.
__global__ void moe_gemm1_kernel(const float* __restrict__ h2,
                                 const float* __restrict__ wg,
                                 const float* __restrict__ wu,
                                 const int* __restrict__ tok4slot,
                                 float* __restrict__ hmid) {
  int e = blockIdx.z;
  int m0 = blockIdx.y * BM;
  int n0 = blockIdx.x * BN;
  const float* Bg = wg + (long)e * HID * IDIM;
  const float* Bu = wu + (long)e * HID * IDIM;
  __shared__ __align__(16) bf16_t As[BM][LDST];
  __shared__ __align__(16) bf16_t Bgs[BN][LDST];
  __shared__ __align__(16) bf16_t Bus[BN][LDST];
  __shared__ int toks[BM];
  int tid = threadIdx.x;
  int wid = tid >> 5, lane = tid & 31;
  int wm = (wid >> 2) * 32, wn = (wid & 3) * 16;
  int lrow = lane & 15;
  int kb = (lane < 16) ? 0 : 8;
  if (tid < BM) toks[tid] = tok4slot[e * CAP + m0 + tid];
  __syncthreads();

  int ar = tid >> 3;
  int ac = (tid & 7) * 4;
  int tk0 = toks[ar], tk1 = toks[ar + 32];
  bool va0 = tk0 >= 0, va1 = tk1 >= 0;
  const float* ap0 = h2 + (long)(va0 ? tk0 : 0) * HID + ac;
  const float* ap1 = h2 + (long)(va1 ? tk1 : 0) * HID + ac;
  int bk = (tid >> 4) * 2;
  int bn = (tid & 15) * 4;
  const float* gp0 = Bg + (long)bk * IDIM + n0 + bn;
  const float* gp1 = gp0 + IDIM;
  const float* up0 = Bu + (long)bk * IDIM + n0 + bn;
  const float* up1 = up0 + IDIM;
  const long bstep = (long)BK * IDIM;
  const float4 z4 = {0.f, 0.f, 0.f, 0.f};

  v8f accg0 = {0.f,0.f,0.f,0.f,0.f,0.f,0.f,0.f};
  v8f accg1 = {0.f,0.f,0.f,0.f,0.f,0.f,0.f,0.f};
  v8f accu0 = {0.f,0.f,0.f,0.f,0.f,0.f,0.f,0.f};
  v8f accu1 = {0.f,0.f,0.f,0.f,0.f,0.f,0.f,0.f};

  // pipeline prologue
  float4 a0 = va0 ? *(const float4*)ap0 : z4;
  float4 a1 = va1 ? *(const float4*)ap1 : z4;
  float4 g0 = *(const float4*)gp0;
  float4 g1 = *(const float4*)gp1;
  float4 u0 = *(const float4*)up0;
  float4 u1 = *(const float4*)up1;
  ap0 += BK; ap1 += BK;
  gp0 += bstep; gp1 += bstep; up0 += bstep; up1 += bstep;

  for (int kk = 0; kk < HID; kk += BK) {
    *(v4bf*)&As[ar][ac]      = pack4(a0);
    *(v4bf*)&As[ar + 32][ac] = pack4(a1);
    float ge0[4] = {g0.x, g0.y, g0.z, g0.w};
    float ge1[4] = {g1.x, g1.y, g1.z, g1.w};
    float ue0[4] = {u0.x, u0.y, u0.z, u0.w};
    float ue1[4] = {u1.x, u1.y, u1.z, u1.w};
#pragma unroll
    for (int j = 0; j < 4; ++j) {
      v2bf pg; pg[0] = (bf16_t)ge0[j]; pg[1] = (bf16_t)ge1[j];
      *(v2bf*)&Bgs[bn + j][bk] = pg;
      v2bf pu; pu[0] = (bf16_t)ue0[j]; pu[1] = (bf16_t)ue1[j];
      *(v2bf*)&Bus[bn + j][bk] = pu;
    }
    __syncthreads();
    if (kk + BK < HID) {
      a0 = va0 ? *(const float4*)ap0 : z4;
      a1 = va1 ? *(const float4*)ap1 : z4;
      g0 = *(const float4*)gp0;
      g1 = *(const float4*)gp1;
      u0 = *(const float4*)up0;
      u1 = *(const float4*)up1;
      ap0 += BK; ap1 += BK;
      gp0 += bstep; gp1 += bstep; up0 += bstep; up1 += bstep;
      __builtin_prefetch(gp0, 0, 1);             // tile k+2
      __builtin_prefetch(up0, 0, 1);
    }
    v16bf af0 = frag_from_lds(&As[wm + lrow][0], kb);
    v16bf af1 = frag_from_lds(&As[wm + 16 + lrow][0], kb);
    v16bf bg  = frag_from_lds(&Bgs[wn + lrow][0], kb);
    v16bf bu  = frag_from_lds(&Bus[wn + lrow][0], kb);
    accg0 = wmma_bf16(af0, bg, accg0);
    accg1 = wmma_bf16(af1, bg, accg1);
    accu0 = wmma_bf16(af0, bu, accu0);
    accu1 = wmma_bf16(af1, bu, accu1);
    __syncthreads();
  }
  int cn = n0 + wn + lrow;
  int rb0 = m0 + wm + ((lane < 16) ? 0 : 8);
#pragma unroll
  for (int r = 0; r < 8; ++r) {
    float g0e = accg0[r], u0e = accu0[r];
    float g1e = accg1[r], u1e = accu1[r];
    float h0 = g0e * (1.f / (1.f + __expf(-g0e))) * u0e;
    float h1 = g1e * (1.f / (1.f + __expf(-g1e))) * u1e;
    hmid[((long)e * CAP + rb0 + r) * IDIM + cn]      = h0;
    hmid[((long)e * CAP + rb0 + 16 + r) * IDIM + cn] = h1;
  }
}

// -------------------------------------------------------- weighted combine --
__global__ void combine_kernel(const float* __restrict__ yb,
                               const float* __restrict__ topw,
                               const int* __restrict__ slotv,
                               float* __restrict__ out) {
  long i = (long)blockIdx.x * 256 + threadIdx.x;
  if (i >= (long)T_TOK * HID) return;
  int t = (int)(i / HID);
  int h = (int)(i % HID);
  float acc = 0.f;
#pragma unroll
  for (int kx = 0; kx < KTOP; ++kx) {
    int s = slotv[t * KTOP + kx];
    if (s >= 0) acc += topw[t * KTOP + kx] * yb[(long)s * HID + h];
  }
  out[i] = acc;
}

// ============================================================================
extern "C" void kernel_launch(void* const* d_in, const int* in_sizes, int n_in,
                              void* d_out, int out_size, void* d_ws, size_t ws_size,
                              hipStream_t stream) {
  (void)in_sizes; (void)n_in; (void)out_size; (void)ws_size;
  const float* hidden    = (const float*)d_in[0];
  const int*   positions = (const int*)  d_in[1];
  const float* ln_in_w   = (const float*)d_in[2];
  const float* ln_post_w = (const float*)d_in[3];
  const float* q_norm_w  = (const float*)d_in[4];
  const float* k_norm_w  = (const float*)d_in[5];
  const float* wq        = (const float*)d_in[6];
  const float* wk        = (const float*)d_in[7];
  const float* wv        = (const float*)d_in[8];
  const float* wo        = (const float*)d_in[9];
  const float* gate_w    = (const float*)d_in[10];
  const float* we_gate   = (const float*)d_in[11];
  const float* we_up     = (const float*)d_in[12];
  const float* we_down   = (const float*)d_in[13];

  float* out   = (float*)d_out;
  float* resid = (float*)d_out + (size_t)T_TOK * HID;

  char* ws = (char*)d_ws;
  size_t off = 0;
  auto alloc = [&](size_t bytes) -> void* {
    void* p = ws + off;
    off = (off + bytes + 255) & ~(size_t)255;
    return p;
  };
  float* h        = (float*)alloc(sizeof(float) * T_TOK * HID);
  float* qb       = (float*)alloc(sizeof(float) * T_TOK * NH * HD);
  float* kbuf     = (float*)alloc(sizeof(float) * T_TOK * NKV * HD);
  float* vbuf     = (float*)alloc(sizeof(float) * T_TOK * NKV * HD);
  float* attn     = (float*)alloc(sizeof(float) * T_TOK * NH * HD);
  float* h2       = (float*)alloc(sizeof(float) * T_TOK * HID);
  float* logits   = (float*)alloc(sizeof(float) * T_TOK * NEXP);
  float* topw     = (float*)alloc(sizeof(float) * T_TOK * KTOP);
  int*   topi     = (int*)  alloc(sizeof(int)   * T_TOK * KTOP);
  int*   slotv    = (int*)  alloc(sizeof(int)   * T_TOK * KTOP);
  int*   tok4slot = (int*)  alloc(sizeof(int)   * NEXP * CAP);
  float* hmid     = (float*)alloc(sizeof(float) * (size_t)NEXP * CAP * IDIM);
  float* yb       = (float*)alloc(sizeof(float) * (size_t)NEXP * CAP * HID);

  // 1. input RMSNorm
  rmsnorm_kernel<<<T_TOK, 256, 0, stream>>>(hidden, ln_in_w, h, HID);
  // 2-4. QKV projections (bf16 WMMA GEMMs)
  gemm_kernel<<<dim3((NH * HD) / BN, T_TOK / BM, 1), 256, 0, stream>>>(
      h, wq, nullptr, qb, T_TOK, NH * HD, HID, 0, 0, 0);
  gemm_kernel<<<dim3((NKV * HD) / BN, T_TOK / BM, 1), 256, 0, stream>>>(
      h, wk, nullptr, kbuf, T_TOK, NKV * HD, HID, 0, 0, 0);
  gemm_kernel<<<dim3((NKV * HD) / BN, T_TOK / BM, 1), 256, 0, stream>>>(
      h, wv, nullptr, vbuf, T_TOK, NKV * HD, HID, 0, 0, 0);
  // 5. q/k norm + RoPE
  qknorm_rope_kernel<<<dim3(T_TOK, NH), HD, 0, stream>>>(qb, q_norm_w, positions, NH);
  qknorm_rope_kernel<<<dim3(T_TOK, NKV), HD, 0, stream>>>(kbuf, k_norm_w, positions, NKV);
  // 6. flash attention (WMMA QK^T and PV)
  attn_kernel<<<dim3(NH, T_TOK / 16), 32, 0, stream>>>(qb, kbuf, vbuf, attn);
  // 7. output projection + residual add -> resid (second tuple output)
  gemm_kernel<<<dim3(HID / BN, T_TOK / BM, 1), 256, 0, stream>>>(
      attn, wo, hidden, resid, T_TOK, HID, NH * HD, 0, 0, 0);
  // 8. post-attention RMSNorm
  rmsnorm_kernel<<<T_TOK, 256, 0, stream>>>(resid, ln_post_w, h2, HID);
  // 9. router logits
  gemm_kernel<<<dim3(NEXP / BN, T_TOK / BM, 1), 256, 0, stream>>>(
      h2, gate_w, nullptr, logits, T_TOK, NEXP, HID, 0, 0, 0);
  // 10. top-8 + renormalize
  topk_kernel<<<T_TOK / 256, 256, 0, stream>>>(logits, topw, topi);
  // 11. deterministic capacity-based dispatch
  init_i32_kernel<<<(NEXP * CAP) / 256, 256, 0, stream>>>(tok4slot, NEXP * CAP, -1);
  rank_kernel<<<(T_TOK * KTOP) / 256, 256, 0, stream>>>(topi, slotv, tok4slot);
  // 12. fused gate/up expert GEMM + SiLU*up
  moe_gemm1_kernel<<<dim3(IDIM / BN, CAP / BM, NEXP), 256, 0, stream>>>(
      h2, we_gate, we_up, tok4slot, hmid);
  // 13. batched down projection (grid.z = expert)
  gemm_kernel<<<dim3(HID / BN, CAP / BM, NEXP), 256, 0, stream>>>(
      hmid, we_down, nullptr, yb, CAP, HID, IDIM,
      (long)CAP * IDIM, (long)IDIM * HID, (long)CAP * HID);
  // 14. weighted combine -> out (first tuple output)
  combine_kernel<<<(T_TOK * HID) / 256, 256, 0, stream>>>(yb, topw, slotv, out);
}
